// Model_38268158607398
// MI455X (gfx1250) — matrix operand order repaired and checked by reading, not compiled, not run on hardware
//
#include <hip/hip_runtime.h>
#include <hip/hip_bf16.h>

namespace {

constexpr int kB = 32, kT = 2048, kI = 128, kH = 512, kO = 128;
constexpr int kG4 = 4 * kH;            // 2048 gate rows
constexpr int kNWG = 16;               // persistent workgroups (== max cluster size)
constexpr int kSlice = kH / kNWG;      // 32 hidden units per WG
constexpr int kNCol = 4 * kSlice;      // 128 gate columns per WG
constexpr int kThreads = 512;          // 16 wave32 per WG
constexpr int kHS = kH + 8;            // padded LDS stride (bf16) -> 1040 B rows
constexpr int kXS = kI + 8;

typedef __attribute__((ext_vector_type(16))) __bf16 v16bf;
typedef __attribute__((ext_vector_type(8)))  __bf16 v8bf;
typedef __attribute__((ext_vector_type(8)))  float  v8f;
typedef __attribute__((ext_vector_type(4)))  unsigned int u32x4;
typedef __attribute__((ext_vector_type(8)))  int i32x8;
typedef __attribute__((ext_vector_type(4)))  int i32x4;

// 16-bit A-matrix fragment (16x32 tile). Per ISA layout:
// lanes 0-15 hold K = {koff+0..7, koff+16..23}; lanes 16-31 hold K = {koff+8..15, koff+24..31}.
__device__ __forceinline__ v16bf a_frag(const __bf16* row, int koff, int half) {
  v8bf lo = *(const v8bf*)(row + koff + half * 8);
  v8bf hi = *(const v8bf*)(row + koff + 16 + half * 8);
  return __builtin_shufflevector(lo, hi, 0, 1, 2, 3, 4, 5, 6, 7,
                                 8, 9, 10, 11, 12, 13, 14, 15);
}

// 16-bit B-matrix fragment (32x16 tile), weights stored [N][K] row-major:
// lanes 0-15 hold K = {koff+0..7, koff+16..23}; lanes 16-31 hold K = {koff+8..15, koff+24..31}.
__device__ __forceinline__ v16bf b_frag(const __bf16* colrow, int koff, int half) {
  v8bf lo = *(const v8bf*)(colrow + koff + half * 8);
  v8bf hi = *(const v8bf*)(colrow + koff + 16 + half * 8);
  return __builtin_shufflevector(lo, hi, 0, 1, 2, 3, 4, 5, 6, 7,
                                 8, 9, 10, 11, 12, 13, 14, 15);
}

__device__ __forceinline__ v8f wmma_bf16(v16bf a, v16bf b, v8f c) {
  return __builtin_amdgcn_wmma_f32_16x16x32_bf16(false, a, false, b,
                                                 (short)0, c, false, false);
}

__device__ __forceinline__ float sigmoidf(float x) {
  return 1.0f / (1.0f + __expf(-x));
}
__device__ __forceinline__ float tanh_fast(float x) {
  x = fminf(fmaxf(x, -15.0f), 15.0f);
  float e = __expf(2.0f * x);
  return (e - 1.0f) / (e + 1.0f);
}

// 6-arg clang-23 form: (g0, g1, g2, g3, extra_g, cpol).
__device__ __forceinline__ void tdm_load_to_lds(u32x4 g0, i32x8 g1,
                                                i32x4 g2, i32x4 g3) {
  i32x8 gz = {};
  __builtin_amdgcn_tensor_load_to_lds(g0, g1, g2, g3, gz, 0);
}

// Device-wide step barrier: CDNA5 cluster barrier when launched as a cluster,
// sense-reversing global atomic barrier otherwise (ClusterID==0 => cluster
// barrier is a NOP, so the fallback is load-bearing).
__device__ __forceinline__ void grid_sync(unsigned* cnt, volatile unsigned* gen) {
  __threadfence();          // release: flush h writes to device scope
  __syncthreads();
  if (threadIdx.x == 0) {
    if (__builtin_amdgcn_cluster_id_x() != 0) {
      __builtin_amdgcn_s_cluster_barrier();
    } else {
      unsigned g = *gen;
      if (atomicAdd(cnt, 1u) == (unsigned)(kNWG - 1)) {
        *cnt = 0u;
        __threadfence();
        atomicAdd((unsigned*)gen, 1u);
      } else {
        while (*gen == g) __builtin_amdgcn_s_sleep(2);
      }
    }
  }
  __syncthreads();
  __threadfence();          // acquire side
}

__global__ void __launch_bounds__(256)
lstm_prep(const float* __restrict__ W_ih, const float* __restrict__ W_hh,
          const float* __restrict__ lin_W, const float* __restrict__ b_ih,
          const float* __restrict__ b_hh, const float* __restrict__ h0,
          __bf16* __restrict__ wih_bf, __bf16* __restrict__ whh_bf,
          __bf16* __restrict__ linw_bf, float* __restrict__ bias,
          __bf16* __restrict__ hbuf, unsigned* __restrict__ bar) {
  int i = blockIdx.x * blockDim.x + threadIdx.x;
  if (i < kG4 * kI)  wih_bf[i]  = (__bf16)W_ih[i];
  if (i < kG4 * kH)  whh_bf[i]  = (__bf16)W_hh[i];
  if (i < kO * kH)   linw_bf[i] = (__bf16)lin_W[i];
  if (i < kG4)       bias[i]    = b_ih[i] + b_hh[i];
  if (i < kB * kH)   hbuf[i]    = (__bf16)h0[i];    // parity-0 h buffer
  if (i < 2)         bar[i]     = 0u;               // barrier count / generation
}

__global__ void __launch_bounds__(kThreads)
lstm_recurrent(const float* __restrict__ data, const float* __restrict__ c0,
               const __bf16* __restrict__ wih_bf, const __bf16* __restrict__ whh_bf,
               const float* __restrict__ bias, __bf16* __restrict__ hbuf,
               __bf16* __restrict__ csbuf, float* __restrict__ out,
               unsigned* __restrict__ bar) {
  __shared__ __bf16 h_s[kB * kHS];        // 33,280 B  TDM destination (padded rows)
  __shared__ __bf16 x_s[kB * kXS];        //  8,704 B  staged x_t (bf16)
  __shared__ float  gates_s[kB * kNCol];  // 16,384 B  this WG's 128 gate columns
  __shared__ float  c_s[kB * kSlice];     //  4,096 B  persistent cell slice

  const int wg   = blockIdx.x;            // hidden slice [wg*32, wg*32+32)
  const int tid  = threadIdx.x;
  const int lane = tid & 31;
  const int wv   = tid >> 5;              // 16 waves -> 16 output tiles (2 M x 8 N)
  const int half = lane >> 4;
  const int l15  = lane & 15;
  const int mt   = wv & 1;                // batch tile
  const int nt   = wv >> 1;               // gate-column tile

  const int cSlice = nt * 16 + l15;                 // 0..127
  const int gGate  = cSlice >> 5;                   // 0..3 (i,f,g,o)
  const int jB     = cSlice & 31;
  const int wrow   = gGate * kH + wg * kSlice + jB; // row in [4H]

  for (int p = tid; p < kB * kSlice; p += kThreads) {
    int b = p >> 5, jl = p & 31;
    c_s[p] = c0[b * kH + wg * kSlice + jl];
  }

  // ---- Tensor Data Mover descriptors: hbuf[par] (32 KB) -> h_s with LDS padding.
  // 1-D tile of 8192 dwords; pad 4 DW after every 256 DW => LDS row stride 1040 B.
  const unsigned ldsH = (unsigned)(size_t)(&h_s[0]);
  i32x8 g1;
  g1[0] = 0x07D20000;       // data_size=4B | pad_enable | pad_interval=256DW | pad_amount=4DW
  g1[1] = 0x20000000;       // tensor_dim0[15:0]=8192 in [63:48]
  g1[2] = 0x00010000;       // tensor_dim1=1
  g1[3] = 0x20000000;       // tile_dim0=8192 in [127:112]
  g1[4] = 0;                // tile_dim1/2 unused
  g1[5] = 8192;             // tensor_dim0_stride
  g1[6] = 0; g1[7] = 0;
  const i32x4 g2 = {1, 1, 0, 0};          // tensor_dim2=1, tensor_dim3=1 (benign)
  const i32x4 g3 = {0, 1 << 16, 0, 0};    // tensor_dim4=1 (benign)
  u32x4 g0p[2];
  {
    unsigned long long ga = (unsigned long long)(size_t)hbuf;
    for (int p = 0; p < 2; ++p) {
      unsigned long long a = ga + (unsigned long long)p * (kB * kH * 2);
      g0p[p][0] = 1u;                                   // count=1 valid descriptor
      g0p[p][1] = ldsH;                                 // LDS destination
      g0p[p][2] = (unsigned)a;                          // global_addr[31:0]
      g0p[p][3] = ((unsigned)(a >> 32) & 0x01FFFFFFu) | 0x80000000u; // addr[56:32] | type=2
    }
  }

  // Prologue: kick TDM for h(t=0) (hbuf parity 0, written by lstm_prep).
  if (tid < 32) {
    tdm_load_to_lds(g0p[0], g1, g2, g3);
  }

  const size_t hfOff = (size_t)kB * kT * kO;        // h_f after decoded
  const size_t cfOff = hfOff + (size_t)kB * kH;     // c_f after h_f
  const __bf16* browx = wih_bf + (size_t)wrow * kI;
  const __bf16* browh = whh_bf + (size_t)wrow * kH;

  for (int t = 0; t < kT; ++t) {
    const int par = t & 1;
    __bf16* hnext = hbuf + (par ^ 1) * (kB * kH);

    // Stage x_t, converting f32 -> bf16 (small: 16 KB f32 in).
    for (int q = tid; q < kB * kI; q += kThreads) {
      int b = q >> 7, i = q & 127;
      x_s[b * kXS + i] = (__bf16)data[((size_t)b * kT + t) * kI + i];
    }
    if (t + 1 < kT && tid < kB)
      __builtin_prefetch(&data[((size_t)tid * kT + (t + 1)) * kI], 0, 0);
    // Wave 0 issued the TDM; it waits, then the WG barrier publishes h_s to all.
    if (tid < 32) __builtin_amdgcn_s_wait_tensorcnt(0);
    __syncthreads();

    // gates tile = x_t @ W_ih^T (K=128) + h @ W_hh^T (K=512); bf16 WMMA, f32 acc.
    // Two independently-named A buffers ping-pong (no copies => no coalescing),
    // so the reload of a0 overlaps wmma(a1) and vice versa.
    const __bf16* arowx = &x_s[(mt * 16 + l15) * kXS];
    const __bf16* arowh = &h_s[(mt * 16 + l15) * kHS];
    v8f acc = {};
    v16bf a0 = a_frag(arowx, 0, half);
    v16bf a1 = a_frag(arowx, 32, half);

    acc = wmma_bf16(a0, b_frag(browx, 0, half), acc);
    a0 = a_frag(arowx, 64, half);
    acc = wmma_bf16(a1, b_frag(browx, 32, half), acc);
    a1 = a_frag(arowx, 96, half);
    acc = wmma_bf16(a0, b_frag(browx, 64, half), acc);
    a0 = a_frag(arowh, 0, half);
    acc = wmma_bf16(a1, b_frag(browx, 96, half), acc);
    a1 = a_frag(arowh, 32, half);
#pragma unroll
    for (int ks = 0; ks < kH / 32; ks += 2) {
      acc = wmma_bf16(a0, b_frag(browh, ks * 32, half), acc);
      if (ks + 2 < kH / 32) a0 = a_frag(arowh, (ks + 2) * 32, half);
      acc = wmma_bf16(a1, b_frag(browh, (ks + 1) * 32, half), acc);
      if (ks + 3 < kH / 32) a1 = a_frag(arowh, (ks + 3) * 32, half);
    }
    // D layout: VGPR r -> M = r + half*8; N = lane&15.
#pragma unroll
    for (int r = 0; r < 8; ++r) {
      int brow = mt * 16 + r + half * 8;
      gates_s[brow * kNCol + cSlice] = acc[r];
    }
    __syncthreads();

    // Fused LSTM cell update for this WG's 32 hidden units x 32 batches.
    for (int p = tid; p < kB * kSlice; p += kThreads) {
      int b = p >> 5, jl = p & 31;
      int c = wg * kSlice + jl;
      float ig = gates_s[b * kNCol + jl]       + bias[c];
      float fg = gates_s[b * kNCol + 32 + jl]  + bias[kH + c];
      float gg = gates_s[b * kNCol + 64 + jl]  + bias[2 * kH + c];
      float og = gates_s[b * kNCol + 96 + jl]  + bias[3 * kH + c];
      float cn = sigmoidf(fg) * c_s[p] + sigmoidf(ig) * tanh_fast(gg);
      float hn = sigmoidf(og) * tanh_fast(cn);
      c_s[p] = cn;
      csbuf[((size_t)b * kT + t) * kH + c] = (__bf16)cn;   // decode input
      hnext[b * kH + c] = (__bf16)hn;
      if (t == kT - 1) {
        out[hfOff + (size_t)b * kH + c] = hn;
        out[cfOff + (size_t)b * kH + c] = cn;
      }
    }
    grid_sync(bar, bar + 1);   // all WGs see new h before anyone DMAs it

    // Kick TDM for next step's h (parity (t+1)&1) while nothing depends on it yet.
    if (t + 1 < kT && tid < 32) {
      tdm_load_to_lds(g0p[(t + 1) & 1], g1, g2, g3);
    }
  }
}

// decoded[65536,128] = cs[65536,512] @ lin_W^T + lin_b ; one 16x16 tile per wave.
__global__ void __launch_bounds__(256)
lstm_decode(const __bf16* __restrict__ csbuf, const __bf16* __restrict__ linw_bf,
            const float* __restrict__ lin_b, float* __restrict__ out) {
  const int tid  = threadIdx.x;
  const int lane = tid & 31;
  const int half = lane >> 4;
  const int l15  = lane & 15;
  const int wid  = blockIdx.x * 8 + (tid >> 5);
  const int rt   = wid >> 3;          // 0..4095 row tile (b*T+t)
  const int nt   = wid & 7;           // 0..7 output-column tile

  const __bf16* arow = csbuf + (size_t)(rt * 16 + l15) * kH;
  const __bf16* brow = linw_bf + (size_t)(nt * 16 + l15) * kH;
  v8f acc = {};
  v16bf a0 = a_frag(arow, 0, half);
  v16bf a1 = a_frag(arow, 32, half);
#pragma unroll
  for (int ks = 0; ks < kH / 32; ks += 2) {
    acc = wmma_bf16(a0, b_frag(brow, ks * 32, half), acc);
    if (ks + 2 < kH / 32) a0 = a_frag(arow, (ks + 2) * 32, half);
    acc = wmma_bf16(a1, b_frag(brow, (ks + 1) * 32, half), acc);
    if (ks + 3 < kH / 32) a1 = a_frag(arow, (ks + 3) * 32, half);
  }
  const int col = nt * 16 + l15;
  const float lb = lin_b[col];
#pragma unroll
  for (int r = 0; r < 8; ++r) {
    int row = rt * 16 + r + half * 8;
    out[(size_t)row * kO + col] = acc[r] + lb;
  }
}

} // anonymous namespace

extern "C" void kernel_launch(void* const* d_in, const int* in_sizes, int n_in,
                              void* d_out, int out_size, void* d_ws, size_t ws_size,
                              hipStream_t stream) {
  const float* data  = (const float*)d_in[0];
  const float* h0    = (const float*)d_in[1];
  const float* c0    = (const float*)d_in[2];
  const float* W_ih  = (const float*)d_in[3];
  const float* W_hh  = (const float*)d_in[4];
  const float* b_ih  = (const float*)d_in[5];
  const float* b_hh  = (const float*)d_in[6];
  const float* lin_W = (const float*)d_in[7];
  const float* lin_b = (const float*)d_in[8];
  float* out = (float*)d_out;

  // Workspace layout (all chunks multiples of 256 B):
  char* w = (char*)d_ws;
  __bf16* wih_bf  = (__bf16*)w;  w += (size_t)kG4 * kI * 2;   // 512 KB
  __bf16* whh_bf  = (__bf16*)w;  w += (size_t)kG4 * kH * 2;   //   2 MB
  __bf16* linw_bf = (__bf16*)w;  w += (size_t)kO * kH * 2;    // 128 KB
  float*  bias    = (float*)w;   w += (size_t)kG4 * 4;        //   8 KB
  __bf16* hbuf    = (__bf16*)w;  w += (size_t)2 * kB * kH * 2;//  64 KB (double-buffered h)
  unsigned* bar   = (unsigned*)w; w += 256;                   // barrier state
  __bf16* csbuf   = (__bf16*)w;  // kB*kT*kH*2 = 64 MB cell-state history (bf16)

  lstm_prep<<<(kG4 * kH + 255) / 256, 256, 0, stream>>>(
      W_ih, W_hh, lin_W, b_ih, b_hh, h0, wih_bf, whh_bf, linw_bf, bias, hbuf, bar);

  lstm_recurrent<<<kNWG, kThreads, 0, stream>>>(
      data, c0, wih_bf, whh_bf, bias, hbuf, csbuf, out, bar);

  lstm_decode<<<(kB * kT) / 16, 256, 0, stream>>>(csbuf, linw_bf, lin_b, out);
}
